// MultiHeadedAttention_58823872086189
// MI455X (gfx1250) — compile-verified, matrix operand order checked by reading, not run on hardware
//
#include <hip/hip_runtime.h>
#include <hip/hip_bf16.h>
#include <math.h>

// ---------------------------------------------------------------------------
// MultiHeadedAttention for MI455X (gfx1250, wave32, WMMA).
//   B=8, C=256, N=2048, H=4, DIM=64.
// - bf16 WMMA (f32 acc) for the two big attention GEMMs (34.4 GFLOP), f32
//   WMMA (16x16x4) for the 1x1-conv GEMMs (precision for BN stats).
// - per-(b,h) 32x2048 softmax strip lives entirely in the 320KB WGP LDS.
// - conv GEMMs stage the shared X tile once per block via
//   global_load_async_to_lds_b128 (ASYNCcnt), double-buffered.
// - BN + LeakyReLU fused into the final GEMM's B-operand load.
// Workspace requirement: ~92 MB.
// ---------------------------------------------------------------------------

#define B_    8
#define C_    256
#define N_    2048
#define H_    4
#define DIM_  64
#define C2_   512
#define EPS_  1e-5f
#define SLOPE_ 0.2f

typedef __attribute__((ext_vector_type(2)))  float  v2f;
typedef __attribute__((ext_vector_type(4)))  float  v4f;
typedef __attribute__((ext_vector_type(8)))  float  v8f;
typedef __attribute__((ext_vector_type(8)))  __bf16 v8bf;
typedef __attribute__((ext_vector_type(16))) __bf16 v16bf;

// --- CDNA5 async global->LDS copy (tracked with ASYNCcnt) ------------------
__device__ __forceinline__ void async_copy_b128_to_lds(void* lds_dst,
                                                       const void* gsrc)
{
    // low 32 bits of the generic shared pointer == LDS byte offset
    const uint32_t l = (uint32_t)(uintptr_t)lds_dst;
    asm volatile("global_load_async_to_lds_b128 %0, %1, off"
                 :: "v"(l), "v"(gsrc) : "memory");
}
__device__ __forceinline__ void wait_async0()
{
    asm volatile("s_wait_asynccnt 0x0" ::: "memory");
}

// pack two contiguous 8-element bf16 chunks into a 16x16x32 A/B fragment
__device__ __forceinline__ v16bf load_bf16_frag(const __bf16* p, int hi)
{
    v8bf lo = *(const v8bf*)(p + hi * 8);
    v8bf hi8 = *(const v8bf*)(p + 16 + hi * 8);
    return __builtin_shufflevector(lo, hi8, 0, 1, 2, 3, 4, 5, 6, 7,
                                            8, 9, 10, 11, 12, 13, 14, 15);
}

// ---------------------------------------------------------------------------
// 1) QKV projections: f32 WMMA GEMM  proj = W @ X + b  per batch, written out
//    as bf16 in attention-friendly layouts (c = d*H + h):
//      Qt,Kt : [b][h][n][d]     Vh : [b][h][d][n]
//    X tile (shared by all 8 waves) staged through LDS with async copies.
// ---------------------------------------------------------------------------
__global__ __launch_bounds__(256)
void proj_qkv_kernel(const float* __restrict__ xq, const float* __restrict__ xk,
                     const float* __restrict__ xv,
                     const float* __restrict__ Wq, const float* __restrict__ bq,
                     const float* __restrict__ Wk, const float* __restrict__ bk,
                     const float* __restrict__ Wv, const float* __restrict__ bv,
                     __bf16* __restrict__ Qt, __bf16* __restrict__ Kt,
                     __bf16* __restrict__ Vh)
{
    const int which = blockIdx.z % 3;           // 0=q 1=k 2=v
    const int b     = blockIdx.z / 3;
    const float* X    = (which == 0) ? xq : (which == 1) ? xk : xv;
    const float* W    = (which == 0) ? Wq : (which == 1) ? Wk : Wv;
    const float* bias = (which == 0) ? bq : (which == 1) ? bk : bv;
    X += (size_t)b * C_ * N_;

    const int tid  = threadIdx.x;
    const int lane = tid & 31;
    const int wave = tid >> 5;
    const int l16  = lane & 15;
    const int hi   = lane >> 4;

    const int o0 = blockIdx.y * 128 + wave * 16;   // output-channel tile
    const int n0 = blockIdx.x * 64;                // position tile

    __shared__ __attribute__((aligned(16))) float xtile[2][4 * 68]; // 68: pad

    const int srow = tid >> 4;          // staging: 64 threads, 16B each
    const int scol = (tid & 15) * 4;

    v8f acc[4];
    #pragma unroll
    for (int t = 0; t < 4; ++t)
        #pragma unroll
        for (int r = 0; r < 8; ++r)
            acc[t][r] = bias[o0 + r + 8 * hi];

    if (tid < 64)
        async_copy_b128_to_lds(&xtile[0][srow * 68 + scol],
                               X + (size_t)srow * N_ + n0 + scol);

    for (int k = 0, it = 0; k < C_; k += 4, ++it) {
        wait_async0();
        __syncthreads();
        const int cur = it & 1;
        if (k + 4 < C_ && tid < 64)
            async_copy_b128_to_lds(&xtile[cur ^ 1][srow * 68 + scol],
                                   X + (size_t)(k + 4 + srow) * N_ + n0 + scol);

        const int ka = k + hi * 2;                 // f32 frag: K = j + 2*hi
        v2f a;
        a.x = W[(size_t)(o0 + l16) * C_ + ka];
        a.y = W[(size_t)(o0 + l16) * C_ + ka + 1];
        const float* xb = &xtile[cur][0];
        #pragma unroll
        for (int t = 0; t < 4; ++t) {
            const int c = t * 16 + l16;
            v2f bf;
            bf.x = xb[(hi * 2) * 68 + c];
            bf.y = xb[(hi * 2 + 1) * 68 + c];
            acc[t] = __builtin_amdgcn_wmma_f32_16x16x4_f32(
                false, a, false, bf, (short)0, acc[t], false, false);
        }
    }

    // branchless scatter: all three layouts share head stride N*DIM
    __bf16* base = ((which == 0) ? Qt : (which == 1) ? Kt : Vh)
                   + (size_t)b * H_ * N_ * DIM_;
    const int sN = (which == 2) ? 1 : DIM_;
    const int sD = (which == 2) ? N_ : 1;
    #pragma unroll
    for (int t = 0; t < 4; ++t) {
        const int n = n0 + t * 16 + l16;
        #pragma unroll
        for (int r = 0; r < 8; ++r) {
            const int o = o0 + r + 8 * hi;
            const int h = o & (H_ - 1);
            const int d = o >> 2;
            base[(size_t)h * N_ * DIM_ + (size_t)n * sN + (size_t)d * sD] =
                (__bf16)acc[t][r];
        }
    }
}

// ---------------------------------------------------------------------------
// 2) scores = (Q^T K)/8 + row softmax.  One block = 32 query rows of one
//    (b,h); the 32x2048 f32 score strip lives in LDS (256KB of 320KB).
// ---------------------------------------------------------------------------
__global__ __launch_bounds__(256)
void attn_scores_kernel(const __bf16* __restrict__ Qt,
                        const __bf16* __restrict__ Kt,
                        float* __restrict__ attn /* [B*H][N][N] */)
{
    extern __shared__ float smem[];
    float* S      = smem;                // [32][N_]
    float* rowmax = S + 32 * N_;         // [32]
    float* rowsum = rowmax + 32;         // [32]
    float* pmax   = rowsum + 32;         // [32][8]
    float* psum   = pmax + 32 * 8;       // [32][8]

    const int bh = blockIdx.y;
    const int n0 = blockIdx.x * 32;
    const __bf16* Q = Qt + (size_t)bh * N_ * DIM_;
    const __bf16* K = Kt + (size_t)bh * N_ * DIM_;

    const int lane = threadIdx.x & 31;
    const int wave = threadIdx.x >> 5;
    const int l16  = lane & 15;
    const int hi   = lane >> 4;
    const int nsub  = (wave & 1) * 16;   // 16-row half of the strip
    const int mwave = wave >> 1;         // 0..3 column stripe

    v16bf afrag[2];
    {
        const __bf16* qrow = Q + (size_t)(n0 + nsub + l16) * DIM_;
        #pragma unroll
        for (int kk = 0; kk < 2; ++kk)
            afrag[kk] = load_bf16_frag(qrow + kk * 32, hi);
    }

    const float scale = 0.125f;          // 1/sqrt(DIM)
    for (int m0 = mwave * 16; m0 < N_; m0 += 64) {
        const __bf16* krow = K + (size_t)(m0 + l16) * DIM_;
        v8f c = {};
        #pragma unroll
        for (int kk = 0; kk < 2; ++kk) {
            v16bf bfrag = load_bf16_frag(krow + kk * 32, hi);
            c = __builtin_amdgcn_wmma_f32_16x16x32_bf16(
                false, afrag[kk], false, bfrag, (short)0, c, false, false);
        }
        #pragma unroll
        for (int r = 0; r < 8; ++r)
            S[(size_t)(nsub + r + 8 * hi) * N_ + m0 + l16] = c[r] * scale;
    }
    __syncthreads();

    // row softmax over the LDS strip: 8 threads per row
    const int row = threadIdx.x >> 3;    // 0..31
    const int tpr = threadIdx.x & 7;     // 0..7
    float lmax = -3.402823466e38f;
    for (int j = tpr; j < N_; j += 8) lmax = fmaxf(lmax, S[(size_t)row * N_ + j]);
    pmax[row * 8 + tpr] = lmax;
    __syncthreads();
    if (tpr == 0) {
        float m = pmax[row * 8];
        #pragma unroll
        for (int i = 1; i < 8; ++i) m = fmaxf(m, pmax[row * 8 + i]);
        rowmax[row] = m;
    }
    __syncthreads();
    const float rm = rowmax[row];
    float lsum = 0.f;
    for (int j = tpr; j < N_; j += 8) lsum += __expf(S[(size_t)row * N_ + j] - rm);
    psum[row * 8 + tpr] = lsum;
    __syncthreads();
    if (tpr == 0) {
        float s = 0.f;
        #pragma unroll
        for (int i = 0; i < 8; ++i) s += psum[row * 8 + i];
        rowsum[row] = 1.0f / s;
    }
    __syncthreads();
    const float inv = rowsum[row];
    float* dst = attn + (size_t)bh * N_ * N_ + (size_t)(n0 + row) * N_;
    for (int j = tpr; j < N_; j += 8)
        dst[j] = __expf(S[(size_t)row * N_ + j] - rm) * inv;
}

// ---------------------------------------------------------------------------
// 3) x_head = attn @ V^T per (b,h).  attn read back f32 (L2-hot), converted
//    to bf16 on load.  Output scattered into x[b][c = d*H + h][n].
// ---------------------------------------------------------------------------
__global__ __launch_bounds__(256)
void attn_v_kernel(const float* __restrict__ attn,
                   const __bf16* __restrict__ Vh,
                   float* __restrict__ x /* [B][C][N] */)
{
    const int bh = blockIdx.y;
    const int b  = bh >> 2;
    const int h  = bh & (H_ - 1);
    const int lane = threadIdx.x & 31;
    const int wave = threadIdx.x >> 5;
    const int l16  = lane & 15;
    const int hi   = lane >> 4;
    const int n0 = blockIdx.x * 128 + wave * 16;

    const float*  A = attn + (size_t)bh * N_ * N_;
    const __bf16* V = Vh + (size_t)bh * DIM_ * N_;

    v8f acc[4] = {};                       // 4 d-tiles cover DIM=64
    for (int m0 = 0; m0 < N_; m0 += 32) {
        const float* arow = A + (size_t)(n0 + l16) * N_ + m0;
        v4f a0 = *(const v4f*)(arow + hi * 8);
        v4f a1 = *(const v4f*)(arow + hi * 8 + 4);
        v4f a2 = *(const v4f*)(arow + 16 + hi * 8);
        v4f a3 = *(const v4f*)(arow + 16 + hi * 8 + 4);
        v16bf af;
        #pragma unroll
        for (int e = 0; e < 4; ++e) {
            af[e]      = (__bf16)a0[e];
            af[4 + e]  = (__bf16)a1[e];
            af[8 + e]  = (__bf16)a2[e];
            af[12 + e] = (__bf16)a3[e];
        }
        #pragma unroll
        for (int t = 0; t < 4; ++t) {
            const __bf16* vrow = V + (size_t)(t * 16 + l16) * N_ + m0;
            v16bf bf = load_bf16_frag(vrow, hi);
            acc[t] = __builtin_amdgcn_wmma_f32_16x16x32_bf16(
                false, af, false, bf, (short)0, acc[t], false, false);
        }
    }
    #pragma unroll
    for (int t = 0; t < 4; ++t)
        #pragma unroll
        for (int r = 0; r < 8; ++r) {
            const int n = n0 + r + 8 * hi;
            const int d = t * 16 + l16;
            const int c = d * H_ + h;
            x[((size_t)b * C_ + c) * N_ + n] = acc[t][r];
        }
}

// ---------------------------------------------------------------------------
// 4) f32-WMMA 1x1-conv GEMM: out = W @ [X0 ; X1] + bias, optional fused
//    BatchNorm + LeakyReLU on the source.  X tile staged once per block via
//    async global->LDS copies (all 8 waves share the same B fragments).
// ---------------------------------------------------------------------------
__global__ __launch_bounds__(256)
void conv_f32_kernel(const float* __restrict__ W, const float* __restrict__ bias,
                     const float* __restrict__ X0, const float* __restrict__ X1,
                     int K, int KX0, int O,
                     const float* __restrict__ bnMean, const float* __restrict__ bnRstd,
                     const float* __restrict__ bnGamma, const float* __restrict__ bnBeta,
                     float* __restrict__ out)
{
    const int b    = blockIdx.z;
    const int tid  = threadIdx.x;
    const int lane = tid & 31;
    const int wave = tid >> 5;
    const int l16  = lane & 15;
    const int hi   = lane >> 4;
    const int o0 = blockIdx.y * 128 + wave * 16;
    const int n0 = blockIdx.x * 64;

    const float* Xb0 = X0 + (size_t)b * KX0 * N_;
    const float* Xb1 = X1 ? (X1 + (size_t)b * (K - KX0) * N_) : nullptr;

    __shared__ __attribute__((aligned(16))) float xtile[2][4 * 68];
    const int srow = tid >> 4;
    const int scol = (tid & 15) * 4;

    auto rowptr = [&](int kk) -> const float* {
        return (kk < KX0) ? (Xb0 + (size_t)kk * N_)
                          : (Xb1 + (size_t)(kk - KX0) * N_);
    };

    v8f acc[4];
    #pragma unroll
    for (int t = 0; t < 4; ++t)
        #pragma unroll
        for (int r = 0; r < 8; ++r)
            acc[t][r] = bias[o0 + r + 8 * hi];

    if (tid < 64)
        async_copy_b128_to_lds(&xtile[0][srow * 68 + scol],
                               rowptr(srow) + n0 + scol);

    for (int k = 0, it = 0; k < K; k += 4, ++it) {
        wait_async0();
        __syncthreads();
        const int cur = it & 1;
        if (k + 4 < K && tid < 64)
            async_copy_b128_to_lds(&xtile[cur ^ 1][srow * 68 + scol],
                                   rowptr(k + 4 + srow) + n0 + scol);

        const int ka = k + hi * 2;
        v2f a;
        a.x = W[(size_t)(o0 + l16) * K + ka];
        a.y = W[(size_t)(o0 + l16) * K + ka + 1];

        float scale0 = 1.f, shift0 = 0.f, scale1 = 1.f, shift1 = 0.f;
        if (bnMean) {
            scale0 = bnRstd[ka] * bnGamma[ka];
            shift0 = bnBeta[ka] - bnMean[ka] * scale0;
            scale1 = bnRstd[ka + 1] * bnGamma[ka + 1];
            shift1 = bnBeta[ka + 1] - bnMean[ka + 1] * scale1;
        }
        const float* xb = &xtile[cur][0];
        #pragma unroll
        for (int t = 0; t < 4; ++t) {
            const int c = t * 16 + l16;
            float x0 = xb[(hi * 2) * 68 + c];
            float x1 = xb[(hi * 2 + 1) * 68 + c];
            if (bnMean) {
                x0 = x0 * scale0 + shift0;
                x1 = x1 * scale1 + shift1;
                x0 = x0 > 0.f ? x0 : SLOPE_ * x0;
                x1 = x1 > 0.f ? x1 : SLOPE_ * x1;
            }
            v2f bf; bf.x = x0; bf.y = x1;
            acc[t] = __builtin_amdgcn_wmma_f32_16x16x4_f32(
                false, a, false, bf, (short)0, acc[t], false, false);
        }
    }

    #pragma unroll
    for (int t = 0; t < 4; ++t) {
        const int n = n0 + t * 16 + l16;
        #pragma unroll
        for (int r = 0; r < 8; ++r)
            out[((size_t)b * O + (o0 + r + 8 * hi)) * N_ + n] = acc[t][r];
    }
}

// ---------------------------------------------------------------------------
// 5) BatchNorm batch statistics: one block per channel, reduce over B*N.
// ---------------------------------------------------------------------------
__global__ __launch_bounds__(256)
void bn_stats_kernel(const float* __restrict__ h,
                     float* __restrict__ mean, float* __restrict__ rstd)
{
    const int c = blockIdx.x;
    const int tid = threadIdx.x;
    float s = 0.f, s2 = 0.f;
    for (int b = 0; b < B_; ++b) {
        const float* p = h + ((size_t)b * C2_ + c) * N_;
        for (int n = tid; n < N_; n += 256) {
            const float v = p[n];
            s += v; s2 += v * v;
        }
    }
    __shared__ float ls[256], ls2[256];
    ls[tid] = s; ls2[tid] = s2;
    __syncthreads();
    for (int off = 128; off > 0; off >>= 1) {
        if (tid < off) { ls[tid] += ls[tid + off]; ls2[tid] += ls2[tid + off]; }
        __syncthreads();
    }
    if (tid == 0) {
        const float inv = 1.0f / (float)(B_ * N_);
        const float mu  = ls[0] * inv;
        const float var = ls2[0] * inv - mu * mu;   // biased, as torch BN
        mean[c] = mu;
        rstd[c] = rsqrtf(var + EPS_);
    }
}

// ---------------------------------------------------------------------------
extern "C" void kernel_launch(void* const* d_in, const int* in_sizes, int n_in,
                              void* d_out, int out_size, void* d_ws, size_t ws_size,
                              hipStream_t stream)
{
    (void)in_sizes; (void)n_in; (void)out_size; (void)ws_size;

    const float* init_query = (const float*)d_in[0];
    const float* key_in     = (const float*)d_in[1];
    const float* value_in   = (const float*)d_in[2];
    const float* Wq = (const float*)d_in[3];  const float* bq = (const float*)d_in[4];
    const float* Wk = (const float*)d_in[5];  const float* bk = (const float*)d_in[6];
    const float* Wv = (const float*)d_in[7];  const float* bv = (const float*)d_in[8];
    const float* Wm = (const float*)d_in[9];  const float* bm = (const float*)d_in[10];
    const float* W1 = (const float*)d_in[11]; const float* b1 = (const float*)d_in[12];
    const float* gamma = (const float*)d_in[13];
    const float* beta  = (const float*)d_in[14];
    const float* W2 = (const float*)d_in[15]; const float* b2 = (const float*)d_in[16];

    float* out  = (float*)d_out;                 // [B][C][N]
    float* attn = out + (size_t)B_ * C_ * N_;    // [B][H][N][N]

    // workspace carve-out (~92 MB)
    char* ws = (char*)d_ws;
    size_t off = 0;
    auto carve = [&](size_t bytes) {
        void* p = ws + off;
        off += (bytes + 255) & ~(size_t)255;
        return p;
    };
    __bf16* Qt   = (__bf16*)carve((size_t)B_ * H_ * N_ * DIM_ * 2);
    __bf16* Kt   = (__bf16*)carve((size_t)B_ * H_ * N_ * DIM_ * 2);
    __bf16* Vh   = (__bf16*)carve((size_t)B_ * H_ * DIM_ * N_ * 2);
    float*  xbuf = (float*) carve((size_t)B_ * C_ * N_ * 4);
    float*  xm   = (float*) carve((size_t)B_ * C_ * N_ * 4);
    float*  hbuf = (float*) carve((size_t)B_ * C2_ * N_ * 4);
    float*  bnMean = (float*)carve(C2_ * 4);
    float*  bnRstd = (float*)carve(C2_ * 4);

    // 1) q/k/v projections (f32 WMMA, async-LDS staged B) -> bf16 head layouts
    proj_qkv_kernel<<<dim3(N_ / 64, C_ / 128, B_ * 3), 256, 0, stream>>>(
        init_query, key_in, value_in, Wq, bq, Wk, bk, Wv, bv, Qt, Kt, Vh);

    // 2) scores + softmax, 256KB LDS strip per block (bf16 WMMA)
    const size_t SHM = ((size_t)32 * N_ + 32 + 32 + 32 * 8 + 32 * 8) * 4;
    hipFuncSetAttribute(reinterpret_cast<const void*>(attn_scores_kernel),
                        hipFuncAttributeMaxDynamicSharedMemorySize, (int)SHM);
    attn_scores_kernel<<<dim3(N_ / 32, B_ * H_), 256, SHM, stream>>>(Qt, Kt, attn);

    // 3) x = attn @ V  (bf16 WMMA, attn converted f32->bf16 on load)
    attn_v_kernel<<<dim3(N_ / 128, B_ * H_), 256, 0, stream>>>(attn, Vh, xbuf);

    // 4) merge conv (Wm)
    conv_f32_kernel<<<dim3(N_ / 64, C_ / 128, B_), 256, 0, stream>>>(
        Wm, bm, xbuf, nullptr, C_, C_, C_,
        nullptr, nullptr, nullptr, nullptr, xm);

    // 5) h = W1 @ concat(xm, init_query) + b1
    conv_f32_kernel<<<dim3(N_ / 64, C2_ / 128, B_), 256, 0, stream>>>(
        W1, b1, xm, init_query, C2_, C_, C2_,
        nullptr, nullptr, nullptr, nullptr, hbuf);

    // 6) batch-norm statistics over (B, N)
    bn_stats_kernel<<<dim3(C2_), 256, 0, stream>>>(hbuf, bnMean, bnRstd);

    // 7) out = W2 @ leakyrelu(BN(h)) + b2   (fused into B-load)
    conv_f32_kernel<<<dim3(N_ / 64, C_ / 128, B_), 256, 0, stream>>>(
        W2, b2, hbuf, nullptr, C2_, C2_, C_,
        bnMean, bnRstd, gamma, beta, out);
}
